// DINAttention_63479616635206
// MI455X (gfx1250) — matrix operand order repaired and checked by reading, not compiled, not run on hardware
//
#include <hip/hip_runtime.h>

typedef __attribute__((ext_vector_type(16))) _Float16 v16h;
typedef __attribute__((ext_vector_type(8)))  _Float16 v8h;
typedef __attribute__((ext_vector_type(8)))  float    v8f;
typedef __attribute__((ext_vector_type(4)))  float    v4f;

constexpr int S  = 200;   // seq len
constexpr int SP = 208;   // padded to 13 * 16
constexpr int E  = 128;   // embed
constexpr int H1 = 128;
constexpr int H2 = 64;
constexpr int K1 = 4 * E; // 512 input feats for layer 1

// ---- LDS layout (bytes), 16B aligned offsets; buffers aliased across phases
constexpr int KH_OFF  = 0;                   // f16 keys [SP][E]      (dead after layer1)
constexpr int QH_OFF  = KH_OFF + SP * E * 2; // f16 q    [E]
constexpr int H1H_OFF = QH_OFF + 256;        // f16 H1   [SP][H1]     (dead after layer2)
constexpr int H2F_OFF = 0;                   // f32 H2   [SP][H2]     aliases KH
constexpr int SC_OFF  = H1H_OFF + SP * H1 * 2; // f32 scores [SP]
constexpr int RED_OFF = SC_OFF + SP * 4;     // f32 reduction scratch [16]
constexpr int LDS_BYTES = RED_OFF + 64;      // = 107,648 bytes

// Repack W1 [512,128] f32 (row-major [in,out]) into WMMA-B fragment order, f16.
// Tile (kb 0..15, n 0..7): 512 halves, lane-major: lane<16 -> N=n*16+lane, K=kb*32+j;
// lane>=16 -> N=n*16+(lane-16), K=kb*32+16+j  (j = 0..15 contiguous halves per lane).
__global__ __launch_bounds__(256) void prep_w1(const float* __restrict__ W1,
                                               _Float16* __restrict__ w1h) {
  int idx    = blockIdx.x * 256 + threadIdx.x;      // < 65536
  int tile   = idx >> 9;
  int within = idx & 511;
  int lane   = within >> 4;
  int j      = within & 15;
  int kb     = tile >> 3, n = tile & 7;
  int klocal = (lane < 16) ? j : 16 + j;
  int K      = kb * 32 + klocal;
  int N      = n * 16 + (lane & 15);
  w1h[idx]   = (_Float16)W1[K * H1 + N];
}

__global__ __launch_bounds__(256) void prep_w2(const float* __restrict__ W2,
                                               _Float16* __restrict__ w2h) {
  int idx    = blockIdx.x * 256 + threadIdx.x;      // < 8192
  int tile   = idx >> 9;
  int within = idx & 511;
  int lane   = within >> 4;
  int j      = within & 15;
  int kb     = tile >> 2, n = tile & 3;
  int klocal = (lane < 16) ? j : 16 + j;
  w2h[idx]   = (_Float16)W2[(kb * 32 + klocal) * H2 + (n * 16 + (lane & 15))];
}

__global__ __launch_bounds__(256) void din_attn_kernel(
    const float* __restrict__ querys, const float* __restrict__ keys,
    const float* __restrict__ b1, const float* __restrict__ b2,
    const float* __restrict__ W3, const float* __restrict__ b3,
    const int* __restrict__ mask,
    const _Float16* __restrict__ w1h, const _Float16* __restrict__ w2h,
    float* __restrict__ out) {
  __shared__ char smem[LDS_BYTES] __attribute__((aligned(16)));
  _Float16* kh  = (_Float16*)(smem + KH_OFF);
  _Float16* qh  = (_Float16*)(smem + QH_OFF);
  _Float16* h1h = (_Float16*)(smem + H1H_OFF);
  float*    h2f = (float*)(smem + H2F_OFF);
  float*    sc  = (float*)(smem + SC_OFF);
  float*    red = (float*)(smem + RED_OFF);

  const int  b    = blockIdx.x;
  const int  tid  = threadIdx.x;
  const int  wave = tid >> 5;
  const int  lane = tid & 31;
  const int  hl   = lane & 15;
  const bool lo   = lane < 16;
  // A-fragment half-group K offsets (contiguous k = 16h + s, same order as w1h/w2h packing)
  const int o0 = lo ? 0 : 16;
  const int o1 = lo ? 8 : 24;

  // ---------------- P0: stage q and keys (f32 -> f16) into LDS ----------------
  if (tid < E) qh[tid] = (_Float16)querys[(size_t)b * E + tid];
  {
    const float* kbase = keys + (size_t)b * S * E;
    for (int i = tid; i < (S * E) / 4; i += 256) {
      v4f v = *(const v4f*)(kbase + i * 4);
      _Float16* dst = kh + i * 4;
      dst[0] = (_Float16)v[0]; dst[1] = (_Float16)v[1];
      dst[2] = (_Float16)v[2]; dst[3] = (_Float16)v[3];
    }
    for (int i = tid; i < (SP - S) * E; i += 256) kh[S * E + i] = (_Float16)0.0f;
  }
  __syncthreads();

  // ---------------- P1: layer1  X[S,512] @ W1 -> relu -> H1 f16 ----------------
  // X columns: [0,128)=q  [128,256)=k  [256,384)=q-k  [384,512)=q*k
  for (int t = wave; t < 13 * 8; t += 8) {
    const int m = t >> 3, n = t & 7;
    const int row = 16 * m + hl;
    const _Float16* krow = kh + row * E;
    v8f c = {};
#pragma unroll
    for (int kb = 0; kb < K1 / 32; ++kb) {
      const int seg = kb >> 2;
      const int cb  = (kb & 3) * 32;
      v8h q0 = *(const v8h*)(qh + cb + o0);
      v8h q1 = *(const v8h*)(qh + cb + o1);
      v8h k0 = *(const v8h*)(krow + cb + o0);
      v8h k1 = *(const v8h*)(krow + cb + o1);
      v8h x0, x1;
      switch (seg) {
        case 0:  x0 = q0;      x1 = q1;      break;
        case 1:  x0 = k0;      x1 = k1;      break;
        case 2:  x0 = q0 - k0; x1 = q1 - k1; break;
        default: x0 = q0 * k0; x1 = q1 * k1; break;
      }
      v16h a = __builtin_shufflevector(x0, x1, 0, 1, 2, 3, 4, 5, 6, 7,
                                       8, 9, 10, 11, 12, 13, 14, 15);
      v16h bm = *(const v16h*)(w1h + ((kb * 8 + n) * 512 + lane * 16));
      c = __builtin_amdgcn_wmma_f32_16x16x32_f16(false, a, false, bm,
                                                 (short)0, c, false, false);
    }
    const int   col  = n * 16 + hl;
    const float bias = b1[col];
#pragma unroll
    for (int g = 0; g < 8; ++g) {
      const int r = 16 * m + (lo ? g : g + 8);
      float v = c[g] + bias;
      h1h[r * H1 + col] = (_Float16)(v > 0.0f ? v : 0.0f);
    }
  }
  __syncthreads();

  // ---------------- P2: layer2  H1[S,128] @ W2 -> relu -> H2 f32 ----------------
  for (int t = wave; t < 13 * 4; t += 8) {
    const int m = t >> 2, n = t & 3;
    const int row = 16 * m + hl;
    const _Float16* hrow = h1h + row * H1;
    v8f c = {};
#pragma unroll
    for (int kb = 0; kb < H1 / 32; ++kb) {
      const int cb = kb * 32;
      v8h x0 = *(const v8h*)(hrow + cb + o0);
      v8h x1 = *(const v8h*)(hrow + cb + o1);
      v16h a = __builtin_shufflevector(x0, x1, 0, 1, 2, 3, 4, 5, 6, 7,
                                       8, 9, 10, 11, 12, 13, 14, 15);
      v16h bm = *(const v16h*)(w2h + ((kb * 4 + n) * 512 + lane * 16));
      c = __builtin_amdgcn_wmma_f32_16x16x32_f16(false, a, false, bm,
                                                 (short)0, c, false, false);
    }
    const int   col  = n * 16 + hl;
    const float bias = b2[col];
#pragma unroll
    for (int g = 0; g < 8; ++g) {
      const int r = 16 * m + (lo ? g : g + 8);
      float v = c[g] + bias;
      h2f[r * H2 + col] = (v > 0.0f ? v : 0.0f);
    }
  }
  __syncthreads();

  // ---------------- P3: layer3 (64-dot) + mask -> scores ----------------
  if (tid < S) {
    float s;
    if (mask[(size_t)b * S + tid] == 0) {
      s = -10000.0f;
    } else {
      s = b3[0];
      const float* hrow = h2f + tid * H2;
#pragma unroll 8
      for (int c2 = 0; c2 < H2; ++c2) s += hrow[c2] * W3[c2];
    }
    sc[tid] = s;
  }
  __syncthreads();

  // ---------------- P4: softmax over S + weighted key sum ----------------
  float v = (tid < S) ? sc[tid] : -3.0e38f;
#pragma unroll
  for (int off = 16; off > 0; off >>= 1) v = fmaxf(v, __shfl_xor(v, off, 32));
  if (lane == 0) red[wave] = v;
  __syncthreads();
  if (tid == 0) {
    float m = red[0];
    for (int i = 1; i < 8; ++i) m = fmaxf(m, red[i]);
    red[8] = m;
  }
  __syncthreads();
  const float smax = red[8];
  float e = (tid < S) ? __expf(sc[tid] - smax) : 0.0f;
  float ssum = e;
#pragma unroll
  for (int off = 16; off > 0; off >>= 1) ssum += __shfl_xor(ssum, off, 32);
  if (lane == 0) red[wave] = ssum;
  __syncthreads();
  if (tid == 0) {
    float s2 = 0.0f;
    for (int i = 0; i < 8; ++i) s2 += red[i];
    red[9] = s2;
  }
  __syncthreads();
  const float inv = 1.0f / red[9];
  if (tid < S) sc[tid] = e * inv;
  __syncthreads();

  if (tid < E) {
    const float* kcol = keys + (size_t)b * S * E + tid;
    float acc = 0.0f;
    for (int s2 = 0; s2 < S; ++s2) acc = fmaf(sc[s2], kcol[s2 * E], acc);
    out[(size_t)b * E + tid] = acc;
  }
}

extern "C" void kernel_launch(void* const* d_in, const int* in_sizes, int n_in,
                              void* d_out, int out_size, void* d_ws, size_t ws_size,
                              hipStream_t stream) {
  const float* querys = (const float*)d_in[0];
  const float* keys   = (const float*)d_in[1];
  const float* W1     = (const float*)d_in[2];
  const float* b1     = (const float*)d_in[3];
  const float* W2     = (const float*)d_in[4];
  const float* b2     = (const float*)d_in[5];
  const float* W3     = (const float*)d_in[6];
  const float* b3     = (const float*)d_in[7];
  const int*   mask   = (const int*)d_in[8];
  float*       out    = (float*)d_out;

  const int B = in_sizes[0] / E;  // querys is [B,1,E]

  _Float16* w1h = (_Float16*)d_ws;
  _Float16* w2h = (_Float16*)((char*)d_ws + (size_t)K1 * H1 * sizeof(_Float16));

  prep_w1<<<(K1 * H1) / 256, 256, 0, stream>>>(W1, w1h);
  prep_w2<<<(H1 * H2) / 256, 256, 0, stream>>>(W2, w2h);
  din_attn_kernel<<<B, 256, 0, stream>>>(querys, keys, b1, b2, W3, b3, mask,
                                         w1h, w2h, out);
}
